// ChEBIRecNN_37915971289924
// MI455X (gfx1250) — matrix operand order repaired and checked by reading, not compiled, not run on hardware
//
#include <hip/hip_runtime.h>

#define NDAGS   16384
#define NNODES  48
#define DFEAT   62
#define MAXP    4
#define DP      64      // padded feature dim (WMMA N)
#define KM      128     // padded merge K (124 -> 128)
#define DPB     16      // DAGs per block (WMMA M)
#define NCLS    500
#define OUTS    72      // padded sOut row stride (shorts): 144B, 16B-aligned
#define XSTR    136     // padded sX row stride (shorts): 272B, 16B-aligned, bank-spread
#define BCH     24      // B-chunk stride (shorts): 16 data + 8 pad, 48B, 16B-aligned

typedef __attribute__((ext_vector_type(16))) __bf16 v16bf;
typedef __attribute__((ext_vector_type(8)))  float  v8f;
typedef __attribute__((ext_vector_type(2)))  float  v2f;
typedef __attribute__((ext_vector_type(2)))  __bf16 v2bf;

union FragBF { v16bf v; unsigned int u[8]; };

static __device__ __forceinline__ unsigned short f2bf_bits(float f) {
  union { __bf16 b; unsigned short u; } cv;
  cv.b = (__bf16)f;                       // native v_cvt, RNE
  return cv.u;
}
static __device__ __forceinline__ unsigned int pack2(float a, float b) {
  union { v2bf b; unsigned int u; } cv;
  v2f x; x[0] = a; x[1] = b;
  cv.b = __builtin_convertvector(x, v2bf); // v_cvt_pk_bf16_f32
  return cv.u;
}
static __device__ __forceinline__ float bf2f_lo(unsigned int u) {
  return __uint_as_float(u << 16);
}
static __device__ __forceinline__ float bf2f_hi(unsigned int u) {
  return __uint_as_float(u & 0xFFFF0000u);
}
// load one 16-short B/A fragment chunk as 2 x ds_load_b128
static __device__ __forceinline__ void loadFrag16(FragBF& f, const unsigned short* base) {
  *(uint4*)&f.u[0] = *(const uint4*)(base);
  *(uint4*)&f.u[4] = *(const uint4*)(base + 8);
}

// ---------------------------------------------------------------------------
// Kernel 1: 16 DAGs per block. Phase 1 seeds outs = relu(atoms@Wsᵀ+b) via
// WMMA; then a 47-step scan with 2 barriers/step:
//   A1: wave w -> DAGs 4w..4w+3: lanes 0-15 score dots (LDS-only),
//       lanes 16-31 stream atom rows into X (overlaps HBM latency)
//   A3: same wave: softmax + agg -> X   (intra-wave LDS RAW, no barrier)
//   B : 16x128x64 bf16 GEMM, wave w owns n-tile w (B frags in registers)
// Sink rows -> d_ws (f32).
// ---------------------------------------------------------------------------
__global__ __launch_bounds__(128) void dag_scan_kernel(
    const float* __restrict__ atoms, const int* __restrict__ preds,
    const float* __restrict__ Wsingle, const float* __restrict__ bsingle,
    const float* __restrict__ Wmerge, const float* __restrict__ bmerge,
    const float* __restrict__ attw, float* __restrict__ lastOut)
{
  extern __shared__ unsigned char smem[];
  unsigned short* sOut = (unsigned short*)smem;          // [DPB*NNODES][OUTS] bf16
  unsigned short* sWm  = sOut + DPB*NNODES*OUTS;         // frag-linear [4ks][2hi][DP][BCH]
  unsigned short* sWs  = sWm + (KM/32)*2*DP*BCH;         // frag-linear [2ks][2hi][DP][BCH]
  unsigned short* sX   = sWs + (DP/32)*2*DP*BCH;         // [DPB][XSTR] bf16
  float* sBs    = (float*)(sX + DPB*XSTR);               // [DP]
  float* sBm    = sBs + DP;                              // [DP]
  float* sAw    = sBm + DP;                              // [DP]
  float* sScore = sAw + DP;                              // [DPB][MAXP]
  int*   sPc    = (int*)(sScore + DPB*MAXP);             // [DPB][MAXP]
  int*   sAny   = sPc + DPB*MAXP;                        // [DPB]
  signed char* sPredsB = (signed char*)(sAny + DPB);     // [DPB][NNODES][MAXP]

  const int tid  = threadIdx.x;
  const int wave = tid >> 5;
  const int lane = tid & 31;
  const int l16  = lane & 15;
  const int hi   = lane >> 4;
  const size_t rowBase = (size_t)blockIdx.x * DPB * NNODES;  // first flat (dag,node) row

  // ---- stage weights, fragment-linear: each lane's 16 B values contiguous --
  for (int i = tid; i < KM * DP; i += 128) {
    int k = i / DP, n = i % DP;
    float v = (k < 2*DFEAT && n < DFEAT) ? Wmerge[n * (2*DFEAT) + k] : 0.0f;
    int ks = k >> 5, hb = (k >> 3) & 1, e = (k & 7) + ((k >> 4) & 1) * 8;
    sWm[((ks*2 + hb)*DP + n)*BCH + e] = f2bf_bits(v);
  }
  for (int i = tid; i < DP * DP; i += 128) {
    int k = i / DP, n = i % DP;
    float v = (k < DFEAT && n < DFEAT) ? Wsingle[n * DFEAT + k] : 0.0f;
    int ks = k >> 5, hb = (k >> 3) & 1, e = (k & 7) + ((k >> 4) & 1) * 8;
    sWs[((ks*2 + hb)*DP + n)*BCH + e] = f2bf_bits(v);
  }
  if (tid < DP) {
    sBs[tid] = (tid < DFEAT) ? bsingle[tid] : 0.0f;
    sBm[tid] = (tid < DFEAT) ? bmerge[tid]  : 0.0f;
    sAw[tid] = (tid < DFEAT) ? attw[tid]    : 0.0f;
  }
  // preds -> LDS as int8 (values in [-1, 47]); off the scan's critical path
  {
    const int* pg = preds + rowBase * MAXP;
    for (int i = tid; i < DPB * NNODES * MAXP; i += 128)
      sPredsB[i] = (signed char)pg[i];
  }
  for (int i = tid; i < DPB * XSTR; i += 128) sX[i] = 0;   // K-pad stays zero
  __syncthreads();

  // ---- phase 1: outs = relu(atoms @ Wsingleᵀ + b) for all 768 rows ----
  for (int mt = wave; mt < (DPB * NNODES) / 16; mt += 4) {
    int lr = mt * 16 + l16;                                // flat local row
    const float* arow = atoms + (rowBase + lr) * DFEAT;
    FragBF fA[2];
#pragma unroll
    for (int ks = 0; ks < 2; ++ks)
#pragma unroll
      for (int v = 0; v < 8; ++v) {
        int k = ks*32 + (v < 4 ? 2*v : 2*v + 8) + hi*8;    // even; pair (k,k+1)
        unsigned int pv = 0;
        if (k + 1 < DFEAT) { float2 a = *(const float2*)(arow + k); pv = pack2(a.x, a.y); }
        fA[ks].u[v] = pv;
      }
#pragma unroll
    for (int nt = 0; nt < 4; ++nt) {
      int n = nt*16 + l16;
      FragBF fB0, fB1;
      loadFrag16(fB0, sWs + ((0*2 + hi)*DP + n)*BCH);
      loadFrag16(fB1, sWs + ((1*2 + hi)*DP + n)*BCH);
      v8f c = {};
      c = __builtin_amdgcn_wmma_f32_16x16x32_bf16(false, fA[0].v, false, fB0.v, (short)0, c, false, false);
      c = __builtin_amdgcn_wmma_f32_16x16x32_bf16(false, fA[1].v, false, fB1.v, (short)0, c, false, false);
      float bias = sBs[n];
#pragma unroll
      for (int r = 0; r < 8; ++r) {
        int lr2 = mt*16 + r + hi*8;                        // flat local row
        float val = c[r] + bias;
        val = val > 0.0f ? val : 0.0f;
        sOut[lr2 * OUTS + n] = f2bf_bits(val);
      }
    }
  }
  __syncthreads();

  // ---- preload W_merge B-fragments for this wave's n-tile (reused 47x) ----
  FragBF fBm[4];
  {
    int n = wave*16 + l16;
#pragma unroll
    for (int ks = 0; ks < 4; ++ks)
      loadFrag16(fBm[ks], sWm + ((ks*2 + hi)*DP + n)*BCH);
  }

  // ---- sequential scan over topo-ordered nodes ----
  const int dgrp = wave * 4;                 // this wave's 4 DAGs
  for (int t = 1; t < NNODES; ++t) {
    // A1: lanes 0-15 score the 4 preds of this wave's DAGs (LDS-only dot,
    //     4 independent accumulators); lanes 16-31 stream atoms -> X[62..123]
    if (lane < 16) {
      int dg = dgrp + (lane >> 2), p = lane & 3;
      int pv = (int)sPredsB[(dg*NNODES + t)*MAXP + p];
      int pc = pv < 0 ? 0 : pv;
      sPc[dg * MAXP + p] = pc;
      float s = -1e30f;
      if (pv >= 0) {
        const unsigned short* row = sOut + (dg * NNODES + pc) * OUTS;  // 16B aligned
        float a0 = 0.0f, a1 = 0.0f, a2 = 0.0f, a3 = 0.0f;
#pragma unroll
        for (int q = 0; q < 7; ++q) {
          uint4 w = *(const uint4*)(row + q*8);
          a0 += bf2f_lo(w.x)*sAw[q*8+0] + bf2f_hi(w.x)*sAw[q*8+1];
          a1 += bf2f_lo(w.y)*sAw[q*8+2] + bf2f_hi(w.y)*sAw[q*8+3];
          a2 += bf2f_lo(w.z)*sAw[q*8+4] + bf2f_hi(w.z)*sAw[q*8+5];
          a3 += bf2f_lo(w.w)*sAw[q*8+6] + bf2f_hi(w.w)*sAw[q*8+7];
        }
        uint2 w2 = *(const uint2*)(row + 56);
        unsigned int w1 = *(const unsigned int*)(row + 60);
        a0 += bf2f_lo(w2.x)*sAw[56] + bf2f_hi(w2.x)*sAw[57];
        a1 += bf2f_lo(w2.y)*sAw[58] + bf2f_hi(w2.y)*sAw[59];
        a2 += bf2f_lo(w1)  *sAw[60] + bf2f_hi(w1)  *sAw[61];
        s = (a0 + a1) + (a2 + a3);
      }
      sScore[dg * MAXP + p] = s;
    } else {
      int l2 = lane - 16;
      int dg = dgrp + (l2 >> 2), q = l2 & 3;
      const float* arow = atoms + ((size_t)(blockIdx.x*DPB + dg) * NNODES + t) * DFEAT;
      unsigned short* xrow = sX + dg * XSTR + DFEAT;
#pragma unroll
      for (int e = 0; e < 8; ++e) {
        int f = q*16 + 2*e;
        if (f < DFEAT) {
          float2 at = *(const float2*)(arow + f);
          *(unsigned int*)(xrow + f) = pack2(at.x, at.y);
        }
      }
      if (l2 == 0 && t + 1 < NNODES) __builtin_prefetch(arow + DFEAT, 0, 1);
    }
    // no barrier: A1 producers and A3 consumers are the same wave (LDS is
    // in-order per wave; compiler inserts s_wait_dscnt)
    // A3: per-DAG softmax (redundant across the 8 threads of a group) +
    //     agg -> X[0..61]
    {
      int dg = dgrp + (lane >> 3), j = lane & 7;
      float s0 = sScore[dg*MAXP+0], s1 = sScore[dg*MAXP+1];
      float s2 = sScore[dg*MAXP+2], s3 = sScore[dg*MAXP+3];
      float mx = fmaxf(fmaxf(s0, s1), fmaxf(s2, s3));
      float e0 = __expf(s0 - mx), e1 = __expf(s1 - mx);
      float e2 = __expf(s2 - mx), e3 = __expf(s3 - mx);
      float inv = 1.0f / (e0 + e1 + e2 + e3);
      float w0 = e0*inv, w1 = e1*inv, w2 = e2*inv, w3 = e3*inv;
      if (j == 0) sAny[dg] = (mx > -1e29f) ? 1 : 0;
      const unsigned short* r0 = sOut + (dg*NNODES + sPc[dg*MAXP+0]) * OUTS;
      const unsigned short* r1 = sOut + (dg*NNODES + sPc[dg*MAXP+1]) * OUTS;
      const unsigned short* r2 = sOut + (dg*NNODES + sPc[dg*MAXP+2]) * OUTS;
      const unsigned short* r3 = sOut + (dg*NNODES + sPc[dg*MAXP+3]) * OUTS;
      unsigned short* xrow = sX + dg * XSTR;
#pragma unroll
      for (int p4 = 0; p4 < 4; ++p4) {
        int f = 2*j + 16*p4;                               // even; pair (f,f+1)
        if (f < DFEAT) {
          unsigned int a0 = *(const unsigned int*)(r0 + f);
          unsigned int a1 = *(const unsigned int*)(r1 + f);
          unsigned int a2 = *(const unsigned int*)(r2 + f);
          unsigned int a3 = *(const unsigned int*)(r3 + f);
          float g0 = w0*bf2f_lo(a0) + w1*bf2f_lo(a1) + w2*bf2f_lo(a2) + w3*bf2f_lo(a3);
          float g1 = w0*bf2f_hi(a0) + w1*bf2f_hi(a1) + w2*bf2f_hi(a2) + w3*bf2f_hi(a3);
          *(unsigned int*)(xrow + f) = pack2(g0, g1);
        }
      }
    }
    __syncthreads();
    // B: merged = relu(X @ Wmergeᵀ + b); wave w owns n-tile w
    {
      FragBF fA[4];
#pragma unroll
      for (int ks = 0; ks < 4; ++ks) {
        // A-fragment = two contiguous 8-short runs -> 2 x ds_load_b128
        *(uint4*)&fA[ks].u[0] = *(const uint4*)(sX + l16*XSTR + ks*32 + hi*8);
        *(uint4*)&fA[ks].u[4] = *(const uint4*)(sX + l16*XSTR + ks*32 + 16 + hi*8);
      }
      v8f c = {};
#pragma unroll
      for (int ks = 0; ks < 4; ++ks)
        c = __builtin_amdgcn_wmma_f32_16x16x32_bf16(false, fA[ks].v, false, fBm[ks].v, (short)0, c, false, false);
      int n = wave*16 + l16;
      float bias = sBm[n];
#pragma unroll
      for (int r = 0; r < 8; ++r) {
        int dg = r + hi*8;
        if (sAny[dg]) {          // keep phase-1 'single' where no predecessors
          float val = c[r] + bias;
          val = val > 0.0f ? val : 0.0f;
          sOut[(dg*NNODES + t) * OUTS + n] = f2bf_bits(val);
        }
      }
    }
    __syncthreads();
  }

  // ---- write sink (node 47) vectors to workspace as f32 ----
  for (int i = tid; i < DPB * DFEAT; i += 128) {
    int dg = i / DFEAT, f = i % DFEAT;
    lastOut[((size_t)blockIdx.x*DPB + dg) * DFEAT + f] =
        bf2f_lo((unsigned int)sOut[(dg*NNODES + (NNODES-1))*OUTS + f]);
  }
}

// ---------------------------------------------------------------------------
// Kernel 2: global softmax pooling over 16384 sink vectors + final layer.
// Single block; 16384 scores cached in LDS.
// ---------------------------------------------------------------------------
__global__ __launch_bounds__(512) void pool_final_kernel(
    const float* __restrict__ last, const float* __restrict__ dagw,
    const float* __restrict__ Wfinal, const float* __restrict__ bfinal,
    float* __restrict__ out)
{
  extern __shared__ unsigned char smem[];
  float* sW      = (float*)smem;        // [NDAGS] scores -> weights
  float* sDw     = sW + NDAGS;          // [DP]
  float* sRed    = sDw + DP;            // [512]
  float* sPart   = sRed + 512;          // [8][DP]
  float* sPooled = sPart + 8 * DP;      // [DP]

  const int tid = threadIdx.x;
  if (tid < DP) sDw[tid] = (tid < DFEAT) ? dagw[tid] : 0.0f;
  __syncthreads();

  float lmax = -1e30f;
  for (int d = tid; d < NDAGS; d += 512) {
    const float* row = last + (size_t)d * DFEAT;
    float s = 0.0f;
    for (int f = 0; f < DFEAT; ++f) s += row[f] * sDw[f];
    sW[d] = s;
    lmax = fmaxf(lmax, s);
  }
  sRed[tid] = lmax;
  __syncthreads();
  for (int off = 256; off > 0; off >>= 1) {
    if (tid < off) sRed[tid] = fmaxf(sRed[tid], sRed[tid + off]);
    __syncthreads();
  }
  float mx = sRed[0];
  __syncthreads();

  float lsum = 0.0f;
  for (int d = tid; d < NDAGS; d += 512) {
    float w = __expf(sW[d] - mx);
    sW[d] = w;
    lsum += w;
  }
  sRed[tid] = lsum;
  __syncthreads();
  for (int off = 256; off > 0; off >>= 1) {
    if (tid < off) sRed[tid] += sRed[tid + off];
    __syncthreads();
  }
  float invZ = 1.0f / sRed[0];

  int f = tid & 63, g = tid >> 6;        // 8 groups x 64 features
  float acc = 0.0f;
  if (f < DFEAT) {
    for (int d = g; d < NDAGS; d += 8) acc += sW[d] * last[(size_t)d * DFEAT + f];
  }
  sPart[g * DP + f] = acc;
  __syncthreads();
  if (tid < DP) {
    float p = 0.0f;
    for (int gg = 0; gg < 8; ++gg) p += sPart[gg * DP + tid];
    sPooled[tid] = (tid < DFEAT) ? p * invZ : 0.0f;
  }
  __syncthreads();

  for (int c = tid; c < NCLS; c += 512) {
    const float* wr = Wfinal + (size_t)c * DFEAT;
    float s = bfinal[c];
    for (int ff = 0; ff < DFEAT; ++ff) s += wr[ff] * sPooled[ff];
    out[c] = 1.0f / (1.0f + __expf(-s));
  }
}

// ---------------------------------------------------------------------------
extern "C" void kernel_launch(void* const* d_in, const int* in_sizes, int n_in,
                              void* d_out, int out_size, void* d_ws, size_t ws_size,
                              hipStream_t stream)
{
  (void)in_sizes; (void)n_in; (void)out_size; (void)ws_size;
  const float* atoms   = (const float*)d_in[0];
  const int*   preds   = (const int*)d_in[1];
  const float* Wsingle = (const float*)d_in[2];
  const float* bsingle = (const float*)d_in[3];
  const float* Wmerge  = (const float*)d_in[4];
  const float* bmerge  = (const float*)d_in[5];
  const float* attw    = (const float*)d_in[6];
  const float* dagw    = (const float*)d_in[7];
  const float* Wfinal  = (const float*)d_in[8];
  const float* bfinal  = (const float*)d_in[9];
  float* out     = (float*)d_out;
  float* lastBuf = (float*)d_ws;   // NDAGS * DFEAT floats (~4.1 MB)

  size_t smem1 = (size_t)(DPB*NNODES*OUTS + (KM/32)*2*DP*BCH + (DP/32)*2*DP*BCH
                          + DPB*XSTR) * 2                                   // bf16 arrays
               + (size_t)(3*DP + DPB*MAXP) * 4                              // float arrays
               + (size_t)(DPB*MAXP + DPB) * 4                               // int arrays
               + (size_t)(DPB*NNODES*MAXP);                                 // int8 preds
  size_t smem2 = (size_t)(NDAGS + DP + 512 + 8*DP + DP) * 4;

  dag_scan_kernel<<<NDAGS / DPB, 128, smem1, stream>>>(
      atoms, preds, Wsingle, bsingle, Wmerge, bmerge, attw, lastBuf);
  pool_final_kernel<<<1, 512, smem2, stream>>>(lastBuf, dagw, Wfinal, bfinal, out);
}
